// LPSC_Cell_62423054680699
// MI455X (gfx1250) — matrix operand order repaired and checked by reading, not compiled, not run on hardware
//
#include <hip/hip_runtime.h>

// MI455X / gfx1250, wave32. D = A(16x32 f16) x B(32x16 f16) + C(16x16 f32).
typedef __attribute__((ext_vector_type(16))) _Float16 v16h;
typedef __attribute__((ext_vector_type(8)))  float    v8f;

#define WAVES 8
#define TILES 2                       // 16-row subtiles per wave
#define ROWS_PER_BLOCK (WAVES * TILES * 16)   // 256

// Wave-local LDS producer->consumer ordering (cross-lane within one wave).
__device__ __forceinline__ void wave_lds_fence() {
  __builtin_amdgcn_wave_barrier();
  asm volatile("s_wait_dscnt 0" ::: "memory");
  __builtin_amdgcn_wave_barrier();
}

union AFrag { unsigned int u[8]; v16h h; };

// A matrix 16x32 f16 (row-major 16x128 tile in LDS, K-step kt selects 32 cols).
// ISA layout: lane<16 holds row m, K in {kh*8..kh*8+7} U {16+kh*8..16+kh*8+7}.
__device__ __forceinline__ v16h load_a_frag(const _Float16* base, int m16, int khalf, int kt) {
  const unsigned int* p = (const unsigned int*)(base + m16 * 128 + kt * 32 + khalf * 8);
  AFrag a;
#pragma unroll
  for (int v = 0; v < 4; ++v) { a.u[v] = p[v]; a.u[4 + v] = p[8 + v]; }
  return a.h;
}

// B matrix 32x16 f16 from swizzled LDS weight image:
// wb[((nt*4+kt)*16 + n)*32 + j] = W[(kt*32 + j)*ldN + nt*16 + n]
// lane reads K = kt*32 + {khalf*8..khalf*8+7} U {16+khalf*8..16+khalf*8+7} at fixed n.
__device__ __forceinline__ v16h load_b_frag(const _Float16* wb, int n16, int khalf, int nt, int kt) {
  const unsigned int* p = (const unsigned int*)(wb + (((nt * 4 + kt) * 16 + n16) * 32) + khalf * 8);
  AFrag b;
#pragma unroll
  for (int v = 0; v < 4; ++v) { b.u[v] = p[v]; b.u[4 + v] = p[8 + v]; }
  return b.h;
}

__global__ __launch_bounds__(256)
void lpsc_kernel(const float* __restrict__ inp, const float* __restrict__ Z,
                 const float* __restrict__ S,   const float* __restrict__ A,
                 const float* __restrict__ Bm,  const float* __restrict__ W0,
                 const float* __restrict__ b0,  const float* __restrict__ W1,
                 const float* __restrict__ b1,  const float* __restrict__ W2,
                 const float* __restrict__ b2,  const float* __restrict__ Wf0,
                 const float* __restrict__ bf0, const float* __restrict__ Wf1,
                 const float* __restrict__ bf1, const float* __restrict__ Wf2,
                 const float* __restrict__ bf2, const int* __restrict__ swp,
                 float* __restrict__ out, int n) {
  __shared__ alignas(16) _Float16 w1h[16384];           // W1 f16, B-frag swizzled
  __shared__ alignas(16) _Float16 wf1s[16384];          // Wf1[idx[k]] f16, swizzled
  __shared__ alignas(16) _Float16 w2h[4096];            // W2 f16, N padded 28->32
  __shared__ alignas(16) _Float16 hA[WAVES][2048];      // per-wave 16x128 activation stage
  __shared__ float nzst[WAVES][32][5];
  __shared__ float xcst[WAVES][32][4];
  __shared__ float kst[WAVES][32][28];
  __shared__ float fvall[WAVES][32][9];
  __shared__ float w0s[640], b0s[128], b1s[128], b2s[32];
  __shared__ float bf1s[128], wf2s[128], bf2s1[1];
  __shared__ float a_s[25], bm_s[10];

  const int tid   = threadIdx.x;
  const int w     = tid >> 5;
  const int lane  = tid & 31;
  const int m16   = lane & 15;
  const int khalf = lane >> 4;
  const int rowbase = blockIdx.x * ROWS_PER_BLOCK + w * (TILES * 16);

  // ---- Phase 0: stage block-shared weights (f32 -> f16, B-frag swizzle) ----
  for (int d = tid; d < 16384; d += 256) {
    int nt = d >> 11, rem = d & 2047, kt = rem >> 9, rem2 = rem & 511;
    int nn = rem2 >> 5, j = rem2 & 31;
    w1h[d] = (_Float16)W1[(kt * 32 + j) * 128 + nt * 16 + nn];
  }
  for (int d = tid; d < 4096; d += 256) {
    int nt = d >> 11, rem = d & 2047, kt = rem >> 9, rem2 = rem & 511;
    int nn = rem2 >> 5, j = rem2 & 31;
    int col = nt * 16 + nn;
    w2h[d] = (col < 28) ? (_Float16)W2[(kt * 32 + j) * 28 + col] : (_Float16)0.f;
  }
  for (int d = tid; d < 640; d += 256) w0s[d] = W0[d];
  if (tid < 128) { b0s[tid] = b0[tid]; b1s[tid] = b1[tid]; }
  if (tid < 32)  b2s[tid] = (tid < 28) ? b2[tid] : 0.f;
  if (tid < 25)  a_s[tid] = A[tid];
  if (tid < 10)  bm_s[tid] = Bm[tid];
  __syncthreads();

  // ---- Per-row scalar state: lanes 0..15 own one row per subtile ----
  float Sv[TILES][9]; float ins[TILES], carb[TILES];
#pragma unroll
  for (int t = 0; t < TILES; ++t) {
    const int r = rowbase + t * 16 + m16;
    if (lane < 16 && r < n) {
#pragma unroll
      for (int j = 0; j < 9; ++j) Sv[t][j] = S[r * 9 + j];
      ins[t] = inp[r * 4 + 0]; carb[t] = inp[r * 4 + 1];
    } else {
#pragma unroll
      for (int j = 0; j < 9; ++j) Sv[t][j] = 0.f;
      ins[t] = 0.f; carb[t] = 0.f;
    }
  }

  // ---- new_Z = Z@A + other@Bmat (160 values per wave, cooperative) ----
  for (int o = lane; o < TILES * 80; o += 32) {
    int m = o / 5, j = o - m * 5;
    int r = rowbase + m;
    float acc = 0.f;
    if (r < n) {
#pragma unroll
      for (int i = 0; i < 5; ++i) acc += Z[r * 5 + i] * a_s[i * 5 + j];
      acc += inp[r * 4 + 2] * bm_s[j] + inp[r * 4 + 3] * bm_s[5 + j];
      out[(size_t)n + (size_t)r * 5 + j] = acc;   // new_Z output
    }
    nzst[w][m][j] = acc;
  }
  wave_lds_fence();

  // ---- Encoder per subtile: layer0 (VALU) -> layer1/2 (WMMA) -> kst ----
#pragma unroll
  for (int t = 0; t < TILES; ++t) {
    {  // layer0 (5->128) -> hA (f16)
      const int colb = lane * 4;
      float w0r[5][4], b0r[4];
#pragma unroll
      for (int q = 0; q < 4; ++q) b0r[q] = b0s[colb + q];
#pragma unroll
      for (int j = 0; j < 5; ++j)
#pragma unroll
        for (int q = 0; q < 4; ++q) w0r[j][q] = w0s[j * 128 + colb + q];
      for (int m = 0; m < 16; ++m) {
        float nz[5];
#pragma unroll
        for (int j = 0; j < 5; ++j) nz[j] = nzst[w][t * 16 + m][j];
        union { _Float16 h[4]; unsigned int u[2]; } pk;
#pragma unroll
        for (int q = 0; q < 4; ++q) {
          float acc = b0r[q];
#pragma unroll
          for (int j = 0; j < 5; ++j) acc += nz[j] * w0r[j][q];
          pk.h[q] = (_Float16)fmaxf(acc, 0.f);
        }
        unsigned int* dst = (unsigned int*)&hA[w][m * 128 + colb];
        dst[0] = pk.u[0]; dst[1] = pk.u[1];
      }
    }
    wave_lds_fence();

    // layer1 (128->128) WMMA, D in regs then written back to hA
    v8f d1[8];
#pragma unroll
    for (int nt = 0; nt < 8; ++nt) {
      v8f c = {};
#pragma unroll
      for (int kt = 0; kt < 4; ++kt) {
        v16h a  = load_a_frag(hA[w], m16, khalf, kt);
        v16h bm = load_b_frag(w1h, m16, khalf, nt, kt);
        c = __builtin_amdgcn_wmma_f32_16x16x32_f16(false, a, false, bm, (short)0, c, false, false);
      }
      d1[nt] = c;
    }
    wave_lds_fence();
#pragma unroll
    for (int nt = 0; nt < 8; ++nt) {
#pragma unroll
      for (int v = 0; v < 8; ++v) {
        int mm = khalf * 8 + v, col = nt * 16 + m16;
        hA[w][mm * 128 + col] = (_Float16)fmaxf(d1[nt][v] + b1s[col], 0.f);
      }
    }
    wave_lds_fence();

    // layer2 (128->28, abs) WMMA -> kst
#pragma unroll
    for (int nt = 0; nt < 2; ++nt) {
      v8f c = {};
#pragma unroll
      for (int kt = 0; kt < 4; ++kt) {
        v16h a  = load_a_frag(hA[w], m16, khalf, kt);
        v16h bm = load_b_frag(w2h, m16, khalf, nt, kt);
        c = __builtin_amdgcn_wmma_f32_16x16x32_f16(false, a, false, bm, (short)0, c, false, false);
      }
#pragma unroll
      for (int v = 0; v < 8; ++v) {
        int col = nt * 16 + m16;
        if (col < 28) kst[w][t * 16 + khalf * 8 + v][col] = fabsf(c[v] + b2s[col]);
      }
    }
    wave_lds_fence();
  }

  // ---- 9 coupled nets: stage Wf1 once per k, run both subtiles per staging ----
#pragma unroll 1
  for (int k = 0; k < 9; ++k) {
    const int id = (k < 8) ? k : 0;          // IDX = [0..7, 0]
    __syncthreads();
    for (int d = tid; d < 16384; d += 256) {
      int nt = d >> 11, rem = d & 2047, kt = rem >> 9, rem2 = rem & 511;
      int nn = rem2 >> 5, j = rem2 & 31;
      wf1s[d] = (_Float16)Wf1[(size_t)id * 16384 + (size_t)(kt * 32 + j) * 128 + nt * 16 + nn];
    }
    if (tid < 128) { bf1s[tid] = bf1[id * 128 + tid]; wf2s[tid] = Wf2[id * 128 + tid]; }
    if (tid == 0)  bf2s1[0] = bf2[id];
    __syncthreads();

    if (k < 8) {  // prefetch next Wf1 slab into GL2 (global_prefetch_b8)
      int idn = (k + 1 < 8) ? (k + 1) : 0;
      const char* p = (const char*)(Wf1 + (size_t)idn * 16384) + tid * 256;
      __builtin_prefetch(p, 0, 0);
      __builtin_prefetch(p + 128, 0, 0);
    }

    // f0 weights for this k (shared by both subtiles)
    const int colb = lane * 4;
    float wr[4][4], br[4];
#pragma unroll
    for (int q = 0; q < 4; ++q) br[q] = bf0[id * 128 + colb + q];
#pragma unroll
    for (int i = 0; i < 4; ++i)
#pragma unroll
      for (int q = 0; q < 4; ++q) wr[i][q] = Wf0[(size_t)id * 512 + i * 128 + colb + q];

#pragma unroll
    for (int t = 0; t < TILES; ++t) {
      // Xc rows for this k, subtile t
      if (lane < 16) {
        float x0, x1, x2 = 0.f, x3 = 0.f;
        switch (k) {
          case 0: x0 = Sv[t][6]; x1 = carb[t]; break;
          case 1: x0 = Sv[t][6]; x1 = Sv[t][7]; break;
          case 2: x0 = Sv[t][7]; x1 = Sv[t][8]; break;
          case 3: x0 = Sv[t][2]; x1 = Sv[t][3]; x2 = ins[t]; break;
          case 4: x0 = Sv[t][3]; x1 = Sv[t][2]; break;
          case 5: x0 = Sv[t][5]; x1 = Sv[t][2]; break;
          case 6: x0 = Sv[t][4]; x1 = Sv[t][2]; break;
          case 7: x0 = Sv[t][5]; x1 = Sv[t][8]; x2 = Sv[t][0]; x3 = Sv[t][1]; break;
          default: x0 = Sv[t][0]; x1 = Sv[t][1]; break;
        }
        xcst[w][t * 16 + m16][0] = x0; xcst[w][t * 16 + m16][1] = x1;
        xcst[w][t * 16 + m16][2] = x2; xcst[w][t * 16 + m16][3] = x3;
      }
      wave_lds_fence();

      // f0: (4 -> 128) VALU -> hA
      for (int m = 0; m < 16; ++m) {
        float x0 = xcst[w][t * 16 + m][0], x1 = xcst[w][t * 16 + m][1];
        float x2 = xcst[w][t * 16 + m][2], x3 = xcst[w][t * 16 + m][3];
        union { _Float16 h[4]; unsigned int u[2]; } pk;
#pragma unroll
        for (int q = 0; q < 4; ++q) {
          float acc = br[q] + x0 * wr[0][q] + x1 * wr[1][q] + x2 * wr[2][q] + x3 * wr[3][q];
          pk.h[q] = (_Float16)fmaxf(acc, 0.f);
        }
        unsigned int* dst = (unsigned int*)&hA[w][m * 128 + colb];
        dst[0] = pk.u[0]; dst[1] = pk.u[1];
      }
      wave_lds_fence();

      // f1: (128 -> 128) WMMA; head (128 -> 1) folded into the epilogue
      float p8[8];
#pragma unroll
      for (int v = 0; v < 8; ++v) p8[v] = 0.f;
#pragma unroll
      for (int nt = 0; nt < 8; ++nt) {
        v8f c = {};
#pragma unroll
        for (int kt = 0; kt < 4; ++kt) {
          v16h a  = load_a_frag(hA[w], m16, khalf, kt);
          v16h bm = load_b_frag(wf1s, m16, khalf, nt, kt);
          c = __builtin_amdgcn_wmma_f32_16x16x32_f16(false, a, false, bm, (short)0, c, false, false);
        }
        const int col = nt * 16 + m16;
        const float wv = wf2s[col];
        const float bv = bf1s[col];
#pragma unroll
        for (int v = 0; v < 8; ++v) p8[v] += fmaxf(c[v] + bv, 0.f) * wv;
      }
#pragma unroll
      for (int v = 0; v < 8; ++v) {   // reduce over the 16 N-lanes of each half
        p8[v] += __shfl_xor(p8[v], 1, 32);
        p8[v] += __shfl_xor(p8[v], 2, 32);
        p8[v] += __shfl_xor(p8[v], 4, 32);
        p8[v] += __shfl_xor(p8[v], 8, 32);
      }
      if (m16 == 0) {                 // lane 0 -> rows 0..7, lane 16 -> rows 8..15
        const float bb2 = bf2s1[0];
#pragma unroll
        for (int v = 0; v < 8; ++v) fvall[w][t * 16 + khalf * 8 + v][k] = p8[v] + bb2;
      }
      wave_lds_fence();
    }
  }
  wave_lds_fence();

  // ---- ODE update + outputs (lanes 0..15, one row per subtile) ----
#pragma unroll
  for (int t = 0; t < TILES; ++t) {
    const int r = rowbase + t * 16 + m16;
    if (lane < 16 && r < n) {
      float Kc[28];
#pragma unroll
      for (int j = 0; j < 28; ++j) Kc[j] = kst[w][t * 16 + m16][j];
      float fv[9];
#pragma unroll
      for (int j = 0; j < 9; ++j) fv[j] = fvall[w][t * 16 + m16][j];
      const float c = (float)swp[0];
      const float Gp = Sv[t][0], Gt = Sv[t][1], Ip = Sv[t][2], Il = Sv[t][3], X = Sv[t][4];
      const float XL = Sv[t][5], Q1 = Sv[t][6], Q2 = Sv[t][7], Qg = Sv[t][8];
      const float k1 = Kc[0], k2 = Kc[1], mm1 = Kc[2], mm2 = Kc[3], mm3 = Kc[4], mm4 = Kc[5];
      const float kgri = Kc[6], kmin = Kc[8], kmax = Kc[9], kabs = Kc[10];
      const float alpha = Kc[11], beta = Kc[12], bb = Kc[13], cc = Kc[14];
      const float D = Kc[15], BW = Kc[16], ff = Kc[17];
      const float kp1 = Kc[18], kp2 = Kc[19], kp3 = Kc[20], ki = Kc[21];
      const float Uii = Kc[22], Vm0 = Kc[23], Vmx = Kc[24], Km0 = Kc[25], r1 = Kc[26], p2u = Kc[27];

      const float EGP  = kp1 - kp2 * Gp - kp3 * XL;
      const float Qsto = Q1 + Q2;
      const float kemptQ = kmin + (kmax - kmin * 0.5f) *
          (tanhf(alpha * (Qsto - bb * D)) - tanhf(beta * (Qsto - cc * D)) + 2.f);
      const float Ra  = ff * kabs * Qg / BW;
      const float Uid = (Vm0 + Vmx * X * r1) * Gt / (Km0 + fabsf(Gt));
      const float DQ1 = -kgri * Q1 + D * carb[t] + c * fv[0];
      const float DQ2 = -kemptQ * Q2 + kgri * Q1 + c * fv[1];
      const float DQg = -kabs * Qg + kemptQ * Q2 + c * fv[2];
      const float DIp = -(mm2 + mm4) * Ip + mm1 * Il + ins[t] + c * fv[3];
      const float DIl = -(mm1 + mm3) * Il + mm2 * Ip + c * fv[4];
      const float DXL = -ki * (XL - Ip) + c * fv[5];
      const float DX  = -p2u * X + p2u * Ip + c * fv[6];
      const float DGp = EGP + Ra - Uii - k1 * Gp + k2 * Gt + c * fv[7];
      const float DGt = -Uid + k1 * Gp - k2 * Gt + c * fv[8];

      out[r] = Gp + DGp;
      const size_t ob = (size_t)n * 6 + (size_t)r * 9;
      const float ns[9] = {Gp + DGp, Gt + DGt, Ip + DIp, Il + DIl, X + DX,
                           XL + DXL, Q1 + DQ1, Q2 + DQ2, Qg + DQg};
#pragma unroll
      for (int j = 0; j < 9; ++j) out[ob + j] = ns[j];
    }
  }
}

extern "C" void kernel_launch(void* const* d_in, const int* in_sizes, int n_in,
                              void* d_out, int out_size, void* d_ws, size_t ws_size,
                              hipStream_t stream) {
  (void)n_in; (void)out_size; (void)d_ws; (void)ws_size;
  const int n = in_sizes[0] / 4;  // batch size
  dim3 grid((n + ROWS_PER_BLOCK - 1) / ROWS_PER_BLOCK), block(256);
  lpsc_kernel<<<grid, block, 0, stream>>>(
      (const float*)d_in[0],  (const float*)d_in[1],  (const float*)d_in[2],
      (const float*)d_in[3],  (const float*)d_in[4],  (const float*)d_in[5],
      (const float*)d_in[6],  (const float*)d_in[7],  (const float*)d_in[8],
      (const float*)d_in[9],  (const float*)d_in[10], (const float*)d_in[11],
      (const float*)d_in[12], (const float*)d_in[13], (const float*)d_in[14],
      (const float*)d_in[15], (const float*)d_in[16], (const int*)d_in[17],
      (float*)d_out, n);
}
